// FGWFModel_49546742726857
// MI455X (gfx1250) — matrix operand order repaired and checked by reading, not compiled, not run on hardware
//
#include <hip/hip_runtime.h>
#include <hip/hip_bf16.h>

// ---------------- problem constants (match reference) ----------------
#define NSZ        1024
#define NN         (1024 * 1024)
#define KATOMS     8
#define OT_LAYERS  3
#define SINK_LAYERS 3
#define GWB_LAYERS 2
#define PROX_BETA  0.1f
#define EPS_T      1e-16f

typedef __attribute__((ext_vector_type(16))) _Float16     v16h;
typedef __attribute__((ext_vector_type(8)))  float        v8f;
typedef __attribute__((ext_vector_type(8)))  unsigned int v8u;

struct __align__(16) U4 { unsigned int x, y, z, w; };
union AccU { v8f v; float f[8]; };

// ---- CDNA5 async global->LDS staging (guarded; falls back cleanly) ----
#ifndef __has_builtin
#define __has_builtin(x) 0
#endif
#if __has_builtin(__builtin_amdgcn_global_load_async_to_lds_b128)
#define USE_ASYNC_LDS 1
// Builtin signature (from hipcc diagnostic): param0 is
// "__attribute__((vector_size(16))) int __device__*"  (AS1 = __device__),
// i.e. typed int4 pointers in global/LDS address spaces.
typedef int v4i_t __attribute__((vector_size(16)));
typedef __attribute__((address_space(1))) v4i_t* as1_v4i;
typedef __attribute__((address_space(3))) v4i_t* as3_v4i;
#else
#define USE_ASYNC_LDS 0
#endif

__device__ __forceinline__ void async_copy16(const _Float16* g, _Float16* l) {
#if USE_ASYNC_LDS
  // memory -> LDS without VGPR round trip; tracked by ASYNCcnt
  __builtin_amdgcn_global_load_async_to_lds_b128(
      (as1_v4i)(unsigned long long)g, (as3_v4i)(unsigned long long)l, 0, 0);
#else
  *(U4*)l = *(const U4*)g;
#endif
}

__device__ __forceinline__ void wait_async_all() {
#if USE_ASYNC_LDS
#if __has_builtin(__builtin_amdgcn_s_wait_asynccnt)
  __builtin_amdgcn_s_wait_asynccnt(0);
#else
  asm volatile("s_wait_asynccnt 0x0" ::: "memory");
#endif
#endif
}

// Build a v16h WMMA fragment from two 16-byte LDS reads.
__device__ __forceinline__ v16h ld_frag(const _Float16* rowp, int o0, int o1) {
  U4 lo = *(const U4*)(rowp + o0);
  U4 hi = *(const U4*)(rowp + o1);
  v8u raw = { lo.x, lo.y, lo.z, lo.w, hi.x, hi.y, hi.z, hi.w };
  return __builtin_bit_cast(v16h, raw);
}

__device__ __forceinline__ float block_reduce_256(float v) {
  __shared__ float red[256];
  red[threadIdx.x] = v;
  __syncthreads();
  for (int s = 128; s > 0; s >>= 1) {
    if ((int)threadIdx.x < s) red[threadIdx.x] += red[threadIdx.x + s];
    __syncthreads();
  }
  float r = red[0];
  __syncthreads();
  return r;
}

// ====================================================================
// WMMA NT GEMM:  C(1024x1024,f32) = scale * A(1024x1024,f16,row) * B^T
// where B is stored row-major as (N x K) f16 (i.e. already transposed).
// Block tile 128x128, K-step 32 (one v_wmma_f32_16x16x32_f16 per frag),
// 8 wave32 waves arranged 4(m) x 2(n), wave tile 32x64 (2x4 accum).
// Double-buffered LDS (async global->LDS staging when available),
// padded row stride 40 halves.
// ====================================================================
__global__ __launch_bounds__(256) void k_gemm_nt(
    float* __restrict__ C, const _Float16* __restrict__ A,
    const _Float16* __restrict__ B,
    const float* __restrict__ scale_ptr, int scale_idx, float scale_mul,
    int accum)
{
  __shared__ _Float16 As[2][128 * 40];
  __shared__ _Float16 Bs[2][128 * 40];

  const int tid  = threadIdx.x;
  const int lane = tid & 31;          // wave32
  const int wave = tid >> 5;          // 0..7
  const int wm   = wave & 3;          // 0..3  (m direction)
  const int wn   = wave >> 2;         // 0..1  (n direction)
  const int l15  = lane & 15;
  const int hig  = lane >> 4;         // half-wave selector
  const int m0   = blockIdx.y * 128;
  const int n0   = blockIdx.x * 128;

  AccU acc[2][4];
  for (int i = 0; i < 2; ++i)
    for (int j = 0; j < 4; ++j)
      for (int r = 0; r < 8; ++r) acc[i][j].f[r] = 0.0f;

  auto stage = [&](int buf, int k0) {
#pragma unroll
    for (int i = 0; i < 2; ++i) {
      int c   = tid + 256 * i;        // 0..511 -> 128 rows x 4 chunks
      int row = c >> 2;
      int kc  = (c & 3) * 8;          // half offset within 32-wide K slab
      async_copy16(A + (size_t)(m0 + row) * NSZ + k0 + kc,
                   &As[buf][row * 40 + kc]);
      async_copy16(B + (size_t)(n0 + row) * NSZ + k0 + kc,
                   &Bs[buf][row * 40 + kc]);
    }
  };

  stage(0, 0);
  wait_async_all();
  __syncthreads();

  for (int kt = 0; kt < NSZ / 32; ++kt) {
    int cur = kt & 1;
    if (kt + 1 < NSZ / 32) stage(cur ^ 1, (kt + 1) * 32);

    const _Float16* as = As[cur];
    const _Float16* bs = Bs[cur];

    v16h af[2], bf[4];
    // A fragment: lanes 0-15 hold K {0..7,16..23}; lanes 16-31 K {8..15,24..31}
#pragma unroll
    for (int mf = 0; mf < 2; ++mf) {
      int row = wm * 32 + mf * 16 + l15;
      int ko  = hig * 8;
      af[mf] = ld_frag(as + row * 40, ko, ko + 16);
    }
    // B fragment: lanes 0-15 hold K {0..7,16..23}; lanes 16-31 K {8..15,24..31}
#pragma unroll
    for (int nf = 0; nf < 4; ++nf) {
      int row = wn * 64 + nf * 16 + l15;
      int kb  = hig * 8;
      bf[nf] = ld_frag(bs + row * 40, kb, kb + 16);
    }

#pragma unroll
    for (int mf = 0; mf < 2; ++mf)
#pragma unroll
      for (int nf = 0; nf < 4; ++nf)
        acc[mf][nf].v = __builtin_amdgcn_wmma_f32_16x16x32_f16(
            false, af[mf], false, bf[nf], (short)0, acc[mf][nf].v,
            false, false);

    wait_async_all();   // drain this wave's next-stage async loads
    __syncthreads();    // all waves' LDS writes visible
  }

  float s = scale_mul;
  if (scale_ptr) s *= scale_ptr[scale_idx];

  // D layout: VGPR r -> (M = r + 8*hig, N = l15) within each 16x16 tile.
#pragma unroll
  for (int mf = 0; mf < 2; ++mf)
    for (int nf = 0; nf < 4; ++nf)
      for (int r = 0; r < 8; ++r) {
        int mrow = m0 + wm * 32 + mf * 16 + r + hig * 8;
        int ncol = n0 + wn * 64 + nf * 16 + l15;
        size_t off = (size_t)mrow * NSZ + ncol;
        float v = s * acc[mf][nf].f[r];
        C[off] = accum ? (C[off] + v) : v;
      }
}

// ---------------- f32 -> f16 convert, optionally also transposed ------
__global__ __launch_bounds__(256) void k_cvt(const float* __restrict__ X,
                                             _Float16* __restrict__ Yn,
                                             _Float16* __restrict__ Yt) {
  __shared__ float tile[32][33];
  int tx = threadIdx.x, ty = threadIdx.y;           // 32 x 8
  int c0 = blockIdx.x * 32, r0 = blockIdx.y * 32;
  for (int i = 0; i < 32; i += 8) {
    float v = X[(size_t)(r0 + ty + i) * NSZ + c0 + tx];
    tile[ty + i][tx] = v;
    if (Yn) Yn[(size_t)(r0 + ty + i) * NSZ + c0 + tx] = (_Float16)v;
  }
  __syncthreads();
  if (Yt)
    for (int i = 0; i < 32; i += 8)
      Yt[(size_t)(c0 + ty + i) * NSZ + r0 + tx] = (_Float16)tile[tx][ty + i];
}

// ---------------- elementwise / reduction helpers --------------------
__global__ void k_sig16(const float* __restrict__ x, _Float16* __restrict__ y,
                        int n) {
  int i = blockIdx.x * 256 + threadIdx.x;
  if (i < n) y[i] = (_Float16)(1.0f / (1.0f + expf(-x[i])));
}

__global__ void k_fill(float* p, float v, int n) {
  int i = blockIdx.x * 256 + threadIdx.x;
  if (i < n) p[i] = v;
}

// out[row] = sum_j f(X[row][j])^2 * w[j]   (f = sigmoid if sigm)
__global__ __launch_bounds__(256) void k_rowsq(const float* __restrict__ X,
                                               const float* __restrict__ w,
                                               float* __restrict__ out,
                                               int sigm) {
  int row = blockIdx.x;
  float acc = 0.0f;
  for (int j = threadIdx.x; j < NSZ; j += 256) {
    float v = X[(size_t)row * NSZ + j];
    if (sigm) v = 1.0f / (1.0f + expf(-v));
    acc += v * v * w[j];
  }
  float tot = block_reduce_256(acc);
  if (threadIdx.x == 0) out[row] = tot;
}

// Kk = exp(-(csr[m]+csc[n]-2S)/beta) * T
__global__ void k_buildK(float* __restrict__ Kk, const float* __restrict__ csr,
                         const float* __restrict__ csc,
                         const float* __restrict__ S,
                         const float* __restrict__ T) {
  int idx = blockIdx.x * 256 + threadIdx.x;
  int m = idx >> 10, n = idx & 1023;
  float cost = csr[m] + csc[n] - 2.0f * S[idx];
  Kk[idx] = expf(-cost * (1.0f / PROX_BETA)) * T[idx];
}

// b[n] = pb[n] / (sum_m Kk[m][n]*a[m] + eps)    (coalesced over n)
__global__ void k_sink_col(float* __restrict__ b, const float* __restrict__ Kk,
                           const float* __restrict__ a,
                           const float* __restrict__ pb) {
  int n = blockIdx.x * 256 + threadIdx.x;
  float acc = 0.0f;
  for (int m = 0; m < NSZ; ++m) acc += Kk[(size_t)m * NSZ + n] * a[m];
  b[n] = pb[n] / (acc + EPS_T);
}

// a[m] = ps[m] / (sum_n Kk[m][n]*b[n] + eps)    (block per row)
__global__ __launch_bounds__(256) void k_sink_row(float* __restrict__ a,
                                                  const float* __restrict__ Kk,
                                                  const float* __restrict__ bv,
                                                  const float* __restrict__ ps) {
  int m = blockIdx.x;
  float acc = 0.0f;
  for (int j = threadIdx.x; j < NSZ; j += 256)
    acc += Kk[(size_t)m * NSZ + j] * bv[j];
  float tot = block_reduce_256(acc);
  if (threadIdx.x == 0) a[m] = ps[m] / (tot + EPS_T);
}

__global__ void k_Tupd(float* __restrict__ T, const float* __restrict__ a,
                       const float* __restrict__ Kk,
                       const float* __restrict__ b) {
  int idx = blockIdx.x * 256 + threadIdx.x;
  int m = idx >> 10, n = idx & 1023;
  T[idx] = a[m] * Kk[idx] * b[n];
}

__global__ void k_outer(float* __restrict__ T, const float* __restrict__ ps,
                        const float* __restrict__ pb) {
  int idx = blockIdx.x * 256 + threadIdx.x;
  int m = idx >> 10, n = idx & 1023;
  T[idx] = ps[m] * pb[n];
}

__global__ void k_divouter(float* __restrict__ G, const float* __restrict__ pb) {
  int idx = blockIdx.x * 256 + threadIdx.x;
  int m = idx >> 10, n = idx & 1023;
  G[idx] = G[idx] / (pb[m] * pb[n]);
}

__global__ __launch_bounds__(256) void k_dist(float* __restrict__ d,
                                              const float* __restrict__ csr,
                                              const float* __restrict__ csc,
                                              const float* __restrict__ S,
                                              const float* __restrict__ T) {
  int idx = blockIdx.x * 256 + threadIdx.x;
  int m = idx >> 10, n = idx & 1023;
  float cost = csr[m] + csc[n] - 2.0f * S[idx];
  float tot = block_reduce_256(cost * T[idx]);
  if (threadIdx.x == 0) atomicAdd(d, tot);
}

// ====================================================================
// Host orchestration
// ====================================================================
extern "C" void kernel_launch(void* const* d_in, const int* in_sizes, int n_in,
                              void* d_out, int out_size, void* d_ws,
                              size_t ws_size, hipStream_t stream) {
  (void)in_sizes; (void)n_in; (void)out_size; (void)ws_size;

  const float* Gm      = (const float*)d_in[0];  // (N,N)
  const float* pm      = (const float*)d_in[1];  // (N)
  const float* atoms   = (const float*)d_in[2];  // (K,N,N) pre-sigmoid
  const float* probs   = (const float*)d_in[3];  // (K,N)
  const float* weights = (const float*)d_in[4];  // (K)
  float*       out     = (float*)d_out;          // [d, G_bary(N*N)]

  // ---- workspace layout (all 256B aligned; ~56 MB total, L2-resident) ----
  char* w = (char*)d_ws;
  auto alloc = [&](size_t bytes) {
    char* p = w;
    w += (bytes + 255) & ~(size_t)255;
    return p;
  };
  _Float16* Gk16   = (_Float16*)alloc((size_t)KATOMS * NN * 2);
  _Float16* Gm16   = (_Float16*)alloc((size_t)NN * 2);
  float*    Gb32   = (float*)   alloc((size_t)NN * 4);
  _Float16* Gb16   = (_Float16*)alloc((size_t)NN * 2);
  float*    GbNew  = (float*)   alloc((size_t)NN * 4);
  float*    T32    = (float*)   alloc((size_t)NN * 4);
  _Float16* T16    = (_Float16*)alloc((size_t)NN * 2);
  _Float16* T16t   = (_Float16*)alloc((size_t)NN * 2);
  float*    Kk32   = (float*)   alloc((size_t)NN * 4);
  float*    U32    = (float*)   alloc((size_t)NN * 4);
  _Float16* U16t   = (_Float16*)alloc((size_t)NN * 2);
  float*    S32    = (float*)   alloc((size_t)NN * 4);
  float*    GT32   = (float*)   alloc((size_t)NN * 4);
  _Float16* GT16t  = (_Float16*)alloc((size_t)NN * 2);
  float*    csr    = (float*)   alloc((size_t)NSZ * 4);
  float*    csc    = (float*)   alloc((size_t)NSZ * 4);
  float*    a_buf  = (float*)   alloc((size_t)NSZ * 4);
  float*    b_buf  = (float*)   alloc((size_t)NSZ * 4);

  const dim3 gGemm(8, 8), bGemm(256);
  const dim3 gCvt(32, 32), bCvt(32, 8);

  auto gemm = [&](float* C, const _Float16* A, const _Float16* B,
                  const float* sp, int si, float sm, int accum) {
    k_gemm_nt<<<gGemm, bGemm, 0, stream>>>(C, A, B, sp, si, sm, accum);
  };
  auto cvt = [&](const float* X, _Float16* Yn, _Float16* Yt) {
    k_cvt<<<gCvt, bCvt, 0, stream>>>(X, Yn, Yt);
  };

  // fgwd proximal-point / Sinkhorn core: leaves transport plan in T32.
  // Uses current barycenter Gb16 + csc; Gs16/csr/ps supplied by caller.
  auto fgwd_core = [&](const _Float16* Gs16, const float* ps,
                       const float* pb) {
    k_outer<<<NN / 256, 256, 0, stream>>>(T32, ps, pb);
    for (int l = 0; l < OT_LAYERS; ++l) {
      cvt(T32, T16, nullptr);
      gemm(U32, T16, Gb16, nullptr, 0, 1.0f, 0);      // U = T * Gb^T
      cvt(U32, nullptr, U16t);
      gemm(S32, Gs16, U16t, nullptr, 0, 1.0f, 0);     // S = Gs * U
      k_buildK<<<NN / 256, 256, 0, stream>>>(Kk32, csr, csc, S32, T32);
      for (int s = 0; s < SINK_LAYERS; ++s) {
        const float* a_in = (s == 0) ? ps : a_buf;
        k_sink_col<<<NSZ / 256, 256, 0, stream>>>(b_buf, Kk32, a_in, pb);
        k_sink_row<<<NSZ, 256, 0, stream>>>(a_buf, Kk32, b_buf, ps);
      }
      k_Tupd<<<NN / 256, 256, 0, stream>>>(T32, a_buf, Kk32, b_buf);
    }
  };

  // ---- one-time conversions ----
  k_sig16<<<(KATOMS * NN) / 256, 256, 0, stream>>>(atoms, Gk16, KATOMS * NN);
  cvt(Gm, Gm16, nullptr);
  (void)hipMemcpyAsync(Gb32, Gm, (size_t)NN * 4, hipMemcpyDeviceToDevice,
                       stream);
  cvt(Gb32, Gb16, nullptr);

  // ---- barycenter alternating optimization ----
  for (int layer = 0; layer < GWB_LAYERS; ++layer) {
    k_rowsq<<<NSZ, 256, 0, stream>>>(Gb32, pm, csc, 0);   // (Gb^2) pb
    k_fill<<<NN / 256, 256, 0, stream>>>(GbNew, 0.0f, NN);
    for (int k = 0; k < KATOMS; ++k) {
      const float* ps = probs + (size_t)k * NSZ;
      k_rowsq<<<NSZ, 256, 0, stream>>>(atoms + (size_t)k * NN, ps, csr, 1);
      fgwd_core(Gk16 + (size_t)k * NN, ps, pm);
      cvt(T32, nullptr, T16t);
      gemm(GT32, Gk16 + (size_t)k * NN, T16t, nullptr, 0, 1.0f, 0); // Gk*T
      cvt(GT32, nullptr, GT16t);
      gemm(GbNew, T16t, GT16t, weights, k, 1.0f, 1);      // += w_k T^T (Gk T)
    }
    k_divouter<<<NN / 256, 256, 0, stream>>>(GbNew, pm);  // / (pb pb^T)
    (void)hipMemcpyAsync(Gb32, GbNew, (size_t)NN * 4, hipMemcpyDeviceToDevice,
                         stream);
    cvt(Gb32, Gb16, nullptr);
  }

  // ---- final fgwd(Gm, pm, G_bary, pm) -> scalar distance ----
  k_rowsq<<<NSZ, 256, 0, stream>>>(Gm, pm, csr, 0);
  k_rowsq<<<NSZ, 256, 0, stream>>>(Gb32, pm, csc, 0);
  fgwd_core(Gm16, pm, pm);
  cvt(T32, T16, nullptr);
  gemm(U32, T16, Gb16, nullptr, 0, 1.0f, 0);
  cvt(U32, nullptr, U16t);
  gemm(S32, Gm16, U16t, nullptr, 0, 1.0f, 0);
  k_fill<<<1, 256, 0, stream>>>(out, 0.0f, 1);
  k_dist<<<NN / 256, 256, 0, stream>>>(out, csr, csc, S32, T32);

  // ---- outputs: [d, G_bary] ----
  (void)hipMemcpyAsync(out + 1, Gb32, (size_t)NN * 4, hipMemcpyDeviceToDevice,
                       stream);
}
